// CharRNN_36859409334429
// MI455X (gfx1250) — matrix operand order repaired and checked by reading, not compiled, not run on hardware
//
#include <hip/hip_runtime.h>
#include <math.h>

typedef __attribute__((ext_vector_type(16))) _Float16 v16h;
typedef __attribute__((ext_vector_type(8)))  float    v8f;

#define Bc 100
#define Tc 50
#define Hc 512
#define Ec 128
#define Vc 8000

__global__ __launch_bounds__(256) void zero_h16(_Float16* p, int n) {
    int i = blockIdx.x * blockDim.x + threadIdx.x;
    if (i < n) p[i] = (_Float16)0.f;
}

__global__ __launch_bounds__(256) void cvt_f16(const float* __restrict__ s,
                                               _Float16* __restrict__ d, int n) {
    int i = blockIdx.x * blockDim.x + threadIdx.x;
    if (i < n) d[i] = (_Float16)s[i];
}

// Pack f32 B[K, Nfull] (row-major) into WMMA-fragment-block f16 layout:
//   dst[((kt*ntot + nt) << 9) + lane*16 + idx]
//   lane = (n%16) + 16*((k%32)>=16), idx = k & 15, kt = k/32, nt = n/16.
__global__ __launch_bounds__(256) void pack_b(const float* __restrict__ src,
                                              _Float16* __restrict__ dst,
                                              int K, int Nfull) {
    int e = blockIdx.x * blockDim.x + threadIdx.x;
    int total = K * Nfull;
    if (e >= total) return;
    int frag = e & 511;
    int blk  = e >> 9;
    int ntot = Nfull >> 4;
    int nt   = blk % ntot;
    int kt   = blk / ntot;
    int lane = frag >> 4, idx = frag & 15;
    int k = (kt << 5) + ((lane >> 4) << 4) + idx;
    int n = (nt << 4) + (lane & 15);
    dst[e] = (_Float16)src[(size_t)k * Nfull + n];
}

// LDS offset of a shared-memory pointer: LDS aperture keeps the in-allocation
// offset in addr[31:0] (ISA 10.2 aperture mapping).
__device__ __forceinline__ unsigned lds_off32(const void* p) {
    return (unsigned)(size_t)p;
}

// Tiled WMMA GEMM, all-f16 operands / f32 accum.
// Block = 256 threads (8 waves). Block tile 32(M) x 128(N), K-step 32.
// Staging uses CDNA5 GLOBAL_LOAD_ASYNC_TO_LDS (ASYNCcnt): A as per-lane
// scattered b32 copies directly into fragment order, B (pre-packed in
// fragment-block layout) as contiguous b128 copies. Double-buffered LDS,
// one s_wait_asynccnt + barrier per K-step.
template<int A_MODE, int EPI>
__global__ __launch_bounds__(256) void gemm_fused(
    const _Float16* __restrict__ A, int lda,
    const int* __restrict__ tokens, const _Float16* __restrict__ embh,
    const _Float16* __restrict__ Bp, int ntot, int nt0,
    int M, int N, int K,
    float* __restrict__ outf, int ldo,
    const float* __restrict__ bias,
    const float* __restrict__ wx, int ldwx,
    const float* __restrict__ alpha, const float* __restrict__ bet1,
    const float* __restrict__ bet2, const float* __restrict__ bvec,
    const _Float16* __restrict__ hbuf, const float* __restrict__ zbuf,
    _Float16* __restrict__ outh, _Float16* __restrict__ outh2, int t)
{
    __shared__ alignas(16) _Float16 ldsA[2][1024];   // 2 x (2 M-subtiles x 512) = 2KB/buf
    __shared__ alignas(16) _Float16 ldsB[2][4096];   // 2 x (8 N-subtiles x 512) = 8KB/buf

    const int tid   = threadIdx.x;
    const int wave  = tid >> 5;
    const int lane  = tid & 31;
    const int mbase = blockIdx.y << 5;
    const int nbase = blockIdx.x << 7;
    const int ntb   = nt0 + (nbase >> 4);

    v8f acc[2] = {{}, {}};
    const int nk = K >> 5;

    // ---- hoist all staging addresses out of the K loop ----
    const _Float16* aga[2];   // global source for A pair i (k advances by 32)
    unsigned        aoff[2];  // LDS dest offset (buffer 0)
    const _Float16* bga[2];   // global source for B chunk i (advances by bstep)
    unsigned        boff[2];  // LDS dest offset (buffer 0)
    #pragma unroll
    for (int i = 0; i < 2; ++i) {
        int e = tid + (i << 8);
        // A: pair (m, even k)
        int m  = e >> 4;
        int k  = (e & 15) << 1;
        int gm = mbase + m;
        int gmc = gm < M ? gm : (M - 1);       // clamp: OOB rows discarded at store
        if constexpr (A_MODE == 0) {
            aga[i] = A + (size_t)gmc * lda + k;
        } else {
            int tt  = gmc / Bc;                // row = t*B + b
            int bb  = gmc - tt * Bc;
            int tok = tokens[bb * Tc + tt];
            aga[i] = embh + ((size_t)tok << 7) + k;
        }
        int half = k >> 4;
        int idx  = k & 15;
        int sub  = m >> 4, mr = m & 15;
        aoff[i]  = lds_off32(&ldsA[0][(sub << 9) + ((mr + (half << 4)) << 4) + idx]);
        // B: 16-byte chunk in fragment-block layout
        int s   = e >> 6;
        int off = (e & 63) << 3;
        int nt  = ntb + s;
        int ntc = nt < ntot ? nt : (ntot - 1); // clamp ragged edge
        bga[i]  = Bp + ((size_t)ntc << 9) + off;
        boff[i] = lds_off32(&ldsB[0][e << 3]);
    }
    const size_t bstep = (size_t)ntot << 9;    // f16 elems per kt step of packed B

    auto stage = [&](int kt, int buf) {
        #pragma unroll
        for (int i = 0; i < 2; ++i) {
            const _Float16* ag = aga[i] + (kt << 5);
            asm volatile("global_load_async_to_lds_b32 %0, %1, off"
                         :: "v"(aoff[i] + (buf << 11)),
                            "v"((unsigned long long)(size_t)ag)
                         : "memory");
        }
        #pragma unroll
        for (int i = 0; i < 2; ++i) {
            const _Float16* bg = bga[i] + (size_t)kt * bstep;
            asm volatile("global_load_async_to_lds_b128 %0, %1, off"
                         :: "v"(boff[i] + (buf << 13)),
                            "v"((unsigned long long)(size_t)bg)
                         : "memory");
        }
    };

    stage(0, 0);
    for (int it = 0; it < nk; ++it) {
        asm volatile("s_wait_asynccnt 0" ::: "memory");  // our async copies landed
        __syncthreads();                                 // everyone's copies visible
        if (it + 1 < nk) stage(it + 1, (it + 1) & 1);
        const int cb = it & 1;
        v16h a0 = *(const v16h*)&ldsA[cb][lane << 4];
        v16h a1 = *(const v16h*)&ldsA[cb][512 + (lane << 4)];
        v16h bv = *(const v16h*)&ldsB[cb][((wave << 5) + lane) << 4];
        acc[0] = __builtin_amdgcn_wmma_f32_16x16x32_f16(
            false, a0, false, bv, (short)0, acc[0], false, false);
        acc[1] = __builtin_amdgcn_wmma_f32_16x16x32_f16(
            false, a1, false, bv, (short)0, acc[1], false, false);
    }

    // ---- fused epilogue (C layout: lane l: col=l%16; VGPR i -> row i+8*(l>=16)) ----
    const int colg = nbase + (wave << 4) + (lane & 15);
    const int rb   = (lane >> 4) << 3;
    #pragma unroll
    for (int sub = 0; sub < 2; ++sub) {
        #pragma unroll
        for (int i = 0; i < 8; ++i) {
            int row = mbase + (sub << 4) + rb + i;
            if (row >= M || colg >= N) continue;
            float v = acc[sub][i];
            if constexpr (EPI == 0) {
                outf[(size_t)row * ldo + colg] = v;
            } else if constexpr (EPI == 1) {
                float wg = wx[(size_t)row * ldwx + colg];
                float g  = alpha[colg] * wg * v + bet1[colg] * v
                         + bet2[colg] * wg + bvec[colg];
                g = 1.f / (1.f + __expf(-g));
                if (colg < Hc)                                   // rh = r*h (f16)
                    outh[(size_t)row * Hc + colg] =
                        (_Float16)(g * (float)hbuf[(size_t)row * Hc + colg]);
                else                                             // z (f32)
                    outf[(size_t)row * Hc + (colg - Hc)] = g;
            } else if constexpr (EPI == 2) {
                int cg   = 2 * Hc + colg;
                float wc = wx[(size_t)row * ldwx + cg];
                float u  = alpha[cg] * wc * v + bet1[cg] * v
                         + bet2[cg] * wc + bvec[cg];
                float c  = tanhf(u);
                float z  = zbuf[(size_t)row * Hc + colg];
                float hn = z * (float)hbuf[(size_t)row * Hc + colg] + (1.f - z) * c;
                outh[(size_t)row * Hc + colg] = (_Float16)hn;    // h_new (f16)
                if (outh2)
                    outh2[(size_t)(row * Tc + t) * Hc + colg] = (_Float16)hn;
            } else {
                outf[(size_t)row * ldo + colg] = v + bias[colg];
            }
        }
    }
}

extern "C" void kernel_launch(void* const* d_in, const int* in_sizes, int n_in,
                              void* d_out, int out_size, void* d_ws, size_t ws_size,
                              hipStream_t stream) {
    const int*   tok  = (const int*)d_in[0];
    const float* emb  = (const float*)d_in[1];
    const float* W0   = (const float*)d_in[2];
    const float* U0   = (const float*)d_in[3];
    const float* b0   = (const float*)d_in[4];
    const float* al0  = (const float*)d_in[5];
    const float* be10 = (const float*)d_in[6];
    const float* be20 = (const float*)d_in[7];
    const float* W1   = (const float*)d_in[8];
    const float* U1   = (const float*)d_in[9];
    const float* b1   = (const float*)d_in[10];
    const float* al1  = (const float*)d_in[11];
    const float* be11 = (const float*)d_in[12];
    const float* be21 = (const float*)d_in[13];
    const float* sw   = (const float*)d_in[14];
    const float* sb   = (const float*)d_in[15];
    float* logits = (float*)d_out;

    // ---- workspace carve: f32 region then f16 region ----
    float* wsf = (float*)d_ws;
    float* WX0 = wsf;                             // (T,B,3H) f32
    float* WX1 = WX0 + (size_t)Tc * Bc * 3 * Hc;  // (B,3H)  f32
    float* ZB  = WX1 + (size_t)Bc * 3 * Hc;       // (B,H)   f32
    _Float16* wsh = (_Float16*)(ZB + (size_t)Bc * Hc);
    _Float16* Hh0  = wsh;                         // (B,H)  f16 (zeroed)
    _Float16* Hh1  = Hh0  + (size_t)Bc * Hc;      // (B,H)  f16 (zeroed)
    _Float16* RHh  = Hh1  + (size_t)Bc * Hc;      // (B,H)  f16
    _Float16* SEQh = RHh  + (size_t)Bc * Hc;      // (B*T,H) f16, row = b*T+t
    _Float16* EMBh = SEQh + (size_t)Tc * Bc * Hc; // (V,E)  f16
    _Float16* W0p  = EMBh + (size_t)Vc * Ec;      // packed (128,1536)
    _Float16* U0p  = W0p  + (size_t)Ec * 3 * Hc;  // packed (512,1536)
    _Float16* W1p  = U0p  + (size_t)Hc * 3 * Hc;  // packed (512,1536)
    _Float16* U1p  = W1p  + (size_t)Hc * 3 * Hc;  // packed (512,1536)
    _Float16* SWp  = U1p  + (size_t)Hc * 3 * Hc;  // packed (512,8000)

    // ---- one-time (per launch) conversions & init ----
    zero_h16<<<(2 * Bc * Hc + 255) / 256, 256, 0, stream>>>(Hh0, 2 * Bc * Hc);
    cvt_f16<<<(Vc * Ec + 255) / 256, 256, 0, stream>>>(emb, EMBh, Vc * Ec);
    pack_b<<<(Ec * 3 * Hc + 255) / 256, 256, 0, stream>>>(W0, W0p, Ec, 3 * Hc);
    pack_b<<<(Hc * 3 * Hc + 255) / 256, 256, 0, stream>>>(U0, U0p, Hc, 3 * Hc);
    pack_b<<<(Hc * 3 * Hc + 255) / 256, 256, 0, stream>>>(W1, W1p, Hc, 3 * Hc);
    pack_b<<<(Hc * 3 * Hc + 255) / 256, 256, 0, stream>>>(U1, U1p, Hc, 3 * Hc);
    pack_b<<<(Hc * Vc + 255) / 256, 256, 0, stream>>>(sw, SWp, Hc, Vc);

    // WX0 = emb[tokens] @ W0   (M=5000, K=128, N=1536)
    gemm_fused<1, 0><<<dim3(12, 157), 256, 0, stream>>>(
        nullptr, 0, tok, EMBh, W0p, 96, 0, Tc * Bc, 3 * Hc, Ec,
        WX0, 3 * Hc, nullptr, nullptr, 0,
        nullptr, nullptr, nullptr, nullptr, nullptr, nullptr, nullptr, nullptr, 0);

    for (int t = 0; t < Tc; ++t) {
        const float* wx0t = WX0 + (size_t)t * Bc * 3 * Hc;
        // layer 0 gates: ug = H0 @ U0[:, :2H] -> RHh (f16), ZB (f32)
        gemm_fused<0, 1><<<dim3(8, 4), 256, 0, stream>>>(
            Hh0, Hc, nullptr, nullptr, U0p, 96, 0, Bc, 2 * Hc, Hc,
            ZB, 0, nullptr, wx0t, 3 * Hc, al0, be10, be20, b0,
            Hh0, nullptr, RHh, nullptr, t);
        // layer 0 candidate: uc = RHh @ U0[:, 2H:] -> Hh0 (in place)
        gemm_fused<0, 2><<<dim3(4, 4), 256, 0, stream>>>(
            RHh, Hc, nullptr, nullptr, U0p, 96, 64, Bc, Hc, Hc,
            nullptr, 0, nullptr, wx0t, 3 * Hc, al0, be10, be20, b0,
            Hh0, ZB, Hh0, nullptr, t);
        // WX1 = H0 @ W1
        gemm_fused<0, 0><<<dim3(12, 4), 256, 0, stream>>>(
            Hh0, Hc, nullptr, nullptr, W1p, 96, 0, Bc, 3 * Hc, Hc,
            WX1, 3 * Hc, nullptr, nullptr, 0,
            nullptr, nullptr, nullptr, nullptr, nullptr, nullptr, nullptr, nullptr, t);
        // layer 1 gates
        gemm_fused<0, 1><<<dim3(8, 4), 256, 0, stream>>>(
            Hh1, Hc, nullptr, nullptr, U1p, 96, 0, Bc, 2 * Hc, Hc,
            ZB, 0, nullptr, WX1, 3 * Hc, al1, be11, be21, b1,
            Hh1, nullptr, RHh, nullptr, t);
        // layer 1 candidate -> Hh1 and SEQh[b*T+t]
        gemm_fused<0, 2><<<dim3(4, 4), 256, 0, stream>>>(
            RHh, Hc, nullptr, nullptr, U1p, 96, 64, Bc, Hc, Hc,
            nullptr, 0, nullptr, WX1, 3 * Hc, al1, be11, be21, b1,
            Hh1, ZB, Hh1, SEQh, t);
    }

    // logits = SEQ @ softmax_w + softmax_b   (M=5000, K=512, N=8000)
    gemm_fused<0, 3><<<dim3((Vc + 127) / 128, (Tc * Bc + 31) / 32), 256, 0, stream>>>(
        SEQh, Hc, nullptr, nullptr, SWp, 500, 0, Tc * Bc, Vc, Hc,
        logits, Vc, sb, nullptr, 0,
        nullptr, nullptr, nullptr, nullptr, nullptr, nullptr, nullptr, nullptr, 0);
}
